// MinibatchTwoBranchGNN_31490700214325
// MI455X (gfx1250) — matrix operand order repaired and checked by reading, not compiled, not run on hardware
//
#include <hip/hip_runtime.h>

// ---------------------------------------------------------------------------
// MinibatchTwoBranchGNN on MI455X (gfx1250, wave32, WMMA).
//
// Dead code removed vs reference: x2/x2b never used; x_mix(hop0) only read as
// [:N2]; b-branch hop-0 edges with dst>=N2 dropped. mean0/mean1 shared.
//
// Pipeline (all on `stream`, graph-capture safe):
//   1) zero sums/counts
//   2) segment-mean aggregations: float4 gathers + global_atomic_add_f32
//   3) fp32 -> bf16(hi)+bf16(lo) split, pre-swizzled into WMMA fragment layout
//   4) dual-A GEMMs  relu(A1@B1 + A2@B2 + b), f32 accum via
//      v_wmma_f32_16x16x32_bf16 (3 WMMAs per K-step for ~fp32 accuracy)
//   5) logits @ Wlin[256,47] + log_softmax (wave-per-row, shfl reductions)
// ---------------------------------------------------------------------------

#define DIN 128
#define DH  256
#define NC  47
#define N0_ 120000
#define N1_ 40000
#define N2_ 10000
#define E0_ 600000
#define E1_ 150000

typedef __attribute__((ext_vector_type(16))) __bf16 v16bf;
typedef __attribute__((ext_vector_type(8)))  float  v8f;

// bf16 conversion via bit ops (RNE) so we don't rely on __bf16 arithmetic.
__device__ __forceinline__ __bf16 f2bf(float f) {
  unsigned u = __builtin_bit_cast(unsigned, f);
  unsigned r = (u + 0x7FFFu + ((u >> 16) & 1u)) >> 16;
  unsigned short h = (unsigned short)r;
  return __builtin_bit_cast(__bf16, h);
}
__device__ __forceinline__ float bf2f(__bf16 b) {
  unsigned u = ((unsigned)__builtin_bit_cast(unsigned short, b)) << 16;
  return __builtin_bit_cast(float, u);
}

__device__ __forceinline__ v8f wmma_bf16(v16bf a, v16bf b, v8f c) {
  // (neg_a, A, neg_b, B, c_mod, C, reuse_a, reuse_b)
  return __builtin_amdgcn_wmma_f32_16x16x32_bf16(false, a, false, b,
                                                 (short)0, c, false, false);
}

// ---------------------------------------------------------------------------
__global__ void k_zero(float* __restrict__ p, long long n) {
  long long i  = (long long)blockIdx.x * blockDim.x + threadIdx.x;
  long long st = (long long)gridDim.x * blockDim.x;
  for (; i < n; i += st) p[i] = 0.0f;
}

// ---------------------------------------------------------------------------
// One wave per edge: gather a 128/256-float row (float4), scatter-add into the
// destination row with global f32 atomics (resident in 192MB L2).
__global__ void k_edge_agg(const float* __restrict__ x,
                           const int* __restrict__ src,
                           const int* __restrict__ dst,
                           const long long* __restrict__ perm,
                           int E, int D, int maxDst,
                           float* __restrict__ sum, float* __restrict__ cnt) {
  int w    = (blockIdx.x << 3) + (threadIdx.x >> 5);
  int lane = threadIdx.x & 31;
  if (w >= E) return;
  int d = dst[w];
  if (d >= maxDst) return;                 // rows never consumed downstream
  long long row = perm ? perm[src[w]] : (long long)src[w];
  const float* xr = x + row * (long long)D;
  float* sr = sum + (long long)d * D;
  for (int c = (lane << 2); c < D; c += 128) {
    float4 v = *(const float4*)(xr + c);
    atomicAdd(sr + c + 0, v.x);
    atomicAdd(sr + c + 1, v.y);
    atomicAdd(sr + c + 2, v.z);
    atomicAdd(sr + c + 3, v.w);
  }
  if (!lane) atomicAdd(cnt + d, 1.0f);
}

// ---------------------------------------------------------------------------
// Build an A-operand in WMMA 16x16x32 bf16 fragment order, split hi/lo.
// A-layout (ISA 7.12.2, 16-bit A 16x32): lane = 16*half + (m&15),
//   j<8 : K = k32 + 8*half + j ;  j>=8 : K = k32 + 16 + 8*half + (j-8)
// mode 0: plain copy    mode 1: mean = sum/max(cnt,1)
// mode 2: mix = m*X[i] + (1-m)*X[idx[i]]
__global__ void k_make_A(const float* __restrict__ src,
                         const float* __restrict__ cnt,
                         const long long* __restrict__ idx,
                         const float* __restrict__ mptr,
                         int M, int K, int mode,
                         __bf16* __restrict__ Ahi, __bf16* __restrict__ Alo) {
  const int nkc = K >> 5;
  long long total = (long long)M * K;
  long long st = (long long)gridDim.x * blockDim.x;
  for (long long t = (long long)blockIdx.x * blockDim.x + threadIdx.x;
       t < total; t += st) {
    int i = (int)(t / K), k = (int)(t % K);
    float v;
    if (mode == 0) {
      v = src[t];
    } else if (mode == 1) {
      v = src[t] / fmaxf(cnt[i], 1.0f);
    } else {
      float m = *mptr;
      v = m * src[t] + (1.0f - m) * src[idx[i] * K + k];
    }
    int mt = i >> 4, mr = i & 15, kc = k >> 5, kl = k & 31;
    int half = (kl >> 3) & 1;
    int j    = (kl & 7) + ((kl >> 4) << 3);
    int lane = (half << 4) | mr;
    long long o = ((((long long)mt * nkc + kc) << 5) + lane) * 16 + j;
    __bf16 h = f2bf(v);
    Ahi[o] = h;
    Alo[o] = f2bf(v - bf2f(h));
  }
}

// B-operand (W is row-major [K,256]).  B-layout: lane = 16*half + (n&15),
//   j<8 : K = k32 + 8*half + j ;  j>=8 : K = k32 + 16 + 8*half + (j-8)
__global__ void k_make_B(const float* __restrict__ W, int K,
                         __bf16* __restrict__ Bhi, __bf16* __restrict__ Blo) {
  const int nkc = K >> 5;
  long long total = (long long)K * DH;
  long long st = (long long)gridDim.x * blockDim.x;
  for (long long t = (long long)blockIdx.x * blockDim.x + threadIdx.x;
       t < total; t += st) {
    int k = (int)(t >> 8), n = (int)(t & 255);
    float v = W[t];
    int nt = n >> 4, nr = n & 15, kc = k >> 5, kl = k & 31;
    int half = (kl >> 3) & 1;
    int j    = (kl & 7) + ((kl >> 4) << 3);
    int lane = (half << 4) | nr;
    long long o = ((((long long)nt * nkc + kc) << 5) + lane) * 16 + j;
    __bf16 h = f2bf(v);
    Bhi[o] = h;
    Blo[o] = f2bf(v - bf2f(h));
  }
}

// ---------------------------------------------------------------------------
// out[M,256] (op)= scale * relu(A1@B1 + A2@B2 + bias)
// scaleSel: 0 -> 1.0, 1 -> *mptr, 2 -> 1-*mptr.  accum: 0 store / 1 +=.
// Block = 256 thr = 8 waves; wave -> 16 rows x 64 cols (4 C-tiles);
// block -> 32 rows x 256 cols.  3 WMMAs per (tile,K32) for bf16x2 accuracy.
__global__ __launch_bounds__(256)
void k_gemm_dual(const __bf16* __restrict__ A1h, const __bf16* __restrict__ A1l,
                 const __bf16* __restrict__ B1h, const __bf16* __restrict__ B1l,
                 const __bf16* __restrict__ A2h, const __bf16* __restrict__ A2l,
                 const __bf16* __restrict__ B2h, const __bf16* __restrict__ B2l,
                 const float* __restrict__ bias, float* __restrict__ out,
                 int M, int nkc, int scaleSel, const float* __restrict__ mptr,
                 int accum) {
  const int lane = threadIdx.x & 31;
  const int wave = threadIdx.x >> 5;
  const int mt = blockIdx.x * 2 + (wave >> 2);   // 16-row tile index
  const int cg = wave & 3;                        // 64-col group
  v8f zero = {0.f, 0.f, 0.f, 0.f, 0.f, 0.f, 0.f, 0.f};
  v8f acc[4] = {zero, zero, zero, zero};

  const __bf16* Ahp[2] = {A1h, A2h};
  const __bf16* Alp[2] = {A1l, A2l};
  const __bf16* Bhp[2] = {B1h, B2h};
  const __bf16* Blp[2] = {B1l, B2l};

  for (int p = 0; p < 2; ++p) {
    const __bf16* Ah = Ahp[p]; const __bf16* Al = Alp[p];
    const __bf16* Bh = Bhp[p]; const __bf16* Bl = Blp[p];
    for (int kc = 0; kc < nkc; ++kc) {
      long long aoff = (((long long)mt * nkc + kc) * 32 + lane) * 16;
      v16bf ah = *(const v16bf*)(Ah + aoff);
      v16bf al = *(const v16bf*)(Al + aoff);
      __builtin_prefetch(Ah + aoff + 512, 0, 1);   // next K-chunk of A
#pragma unroll
      for (int t = 0; t < 4; ++t) {
        long long boff =
            (((long long)(cg * 4 + t) * nkc + kc) * 32 + lane) * 16;
        v16bf bh = *(const v16bf*)(Bh + boff);
        v16bf bl = *(const v16bf*)(Bl + boff);
        acc[t] = wmma_bf16(ah, bh, acc[t]);   // hi*hi
        acc[t] = wmma_bf16(al, bh, acc[t]);   // lo*hi
        acc[t] = wmma_bf16(ah, bl, acc[t]);   // hi*lo
      }
    }
  }

  float s = (scaleSel == 0) ? 1.0f
            : ((scaleSel == 1) ? *mptr : (1.0f - *mptr));
  int half = lane >> 4, n0 = lane & 15;
#pragma unroll
  for (int t = 0; t < 4; ++t) {
    int col = (cg * 4 + t) * 16 + n0;
    float bcol = bias[col];
#pragma unroll
    for (int v = 0; v < 8; ++v) {
      int row = mt * 16 + v + (half << 3);   // C layout: M = v + 8*half
      if (row < M) {
        float r = fmaxf(acc[t][v] + bcol, 0.0f) * s;
        long long o = (long long)row * DH + col;
        if (accum) out[o] += r; else out[o] = r;
      }
    }
  }
}

// ---------------------------------------------------------------------------
// One wave per row: logits[47] = x@Wlin + blin, then log_softmax via shfl.
__global__ void k_logits_lsm(const float* __restrict__ X,
                             const float* __restrict__ Wlin,
                             const float* __restrict__ blin,
                             float* __restrict__ out, int M) {
  int w    = (blockIdx.x << 3) + (threadIdx.x >> 5);
  int lane = threadIdx.x & 31;
  if (w >= M) return;
  const float* xr = X + (long long)w * DH;
  bool v1 = (lane + 32) < NC;
  float a0 = blin[lane];
  float a1 = v1 ? blin[lane + 32] : 0.0f;
  for (int k = 0; k < DH; ++k) {
    float xv = xr[k];
    a0 = fmaf(xv, Wlin[k * NC + lane], a0);
    if (v1) a1 = fmaf(xv, Wlin[k * NC + lane + 32], a1);
  }
  float mx = v1 ? fmaxf(a0, a1) : a0;
  for (int off = 16; off; off >>= 1) mx = fmaxf(mx, __shfl_xor(mx, off));
  float se = __expf(a0 - mx) + (v1 ? __expf(a1 - mx) : 0.0f);
  for (int off = 16; off; off >>= 1) se += __shfl_xor(se, off);
  float lse = mx + __logf(se);
  out[(long long)w * NC + lane] = a0 - lse;
  if (v1) out[(long long)w * NC + lane + 32] = a1 - lse;
}

// ---------------------------------------------------------------------------
extern "C" void kernel_launch(void* const* d_in, const int* in_sizes, int n_in,
                              void* d_out, int out_size, void* d_ws,
                              size_t ws_size, hipStream_t stream) {
  (void)in_sizes; (void)n_in; (void)out_size; (void)ws_size;

  const float*     x0    = (const float*)d_in[0];
  const int*       src0  = (const int*)d_in[1];
  const int*       dst0  = (const int*)d_in[2];
  const int*       src1  = (const int*)d_in[3];
  const int*       dst1  = (const int*)d_in[4];
  const int*       srcb0 = (const int*)d_in[5];
  const int*       dstb0 = (const int*)d_in[6];
  const int*       srcb1 = (const int*)d_in[7];
  const int*       dstb1 = (const int*)d_in[8];
  const long long* idx   = (const long long*)d_in[9];   // int64 per reference
  const float*     mr    = (const float*)d_in[10];
  const float*     Wl0   = (const float*)d_in[11];
  const float*     Wr0   = (const float*)d_in[12];
  const float*     b0    = (const float*)d_in[13];
  const float*     Wl1   = (const float*)d_in[14];
  const float*     Wr1   = (const float*)d_in[15];
  const float*     b1    = (const float*)d_in[16];
  const float*     Wlin  = (const float*)d_in[17];
  const float*     blin  = (const float*)d_in[18];
  float*           out   = (float*)d_out;

  // ---- workspace carve-up (256B aligned) --------------------------------
  char* wp = (char*)d_ws;
  auto alloc = [&](size_t bytes) -> char* {
    char* p = wp; wp += (bytes + 255) & ~(size_t)255; return p;
  };
  const size_t F = sizeof(float);
  const int P2 = 10016;                       // N2 rounded up to 32 rows

  float* sum0  = (float*)alloc((size_t)N1_ * DIN * F);
  float* sumb0 = (float*)alloc((size_t)N2_ * DIN * F);
  float* sum1  = (float*)alloc((size_t)N2_ * DH * F);
  float* sumb1 = (float*)alloc((size_t)N2_ * DH * F);
  float* cnt0  = (float*)alloc((size_t)N1_ * F);
  float* cntb0 = (float*)alloc((size_t)N2_ * F);
  float* cnt1  = (float*)alloc((size_t)N2_ * F);
  float* cntb1 = (float*)alloc((size_t)N2_ * F);
  long long zeroN = (long long)((wp - (char*)d_ws) / 4);

  float* x1  = (float*)alloc((size_t)N1_ * DH * F);
  float* xm1 = (float*)alloc((size_t)N2_ * DH * F);
  float* xm2 = (float*)alloc((size_t)N2_ * DH * F);

  __bf16* M0h  = (__bf16*)alloc((size_t)N1_ * DIN * 2);
  __bf16* M0l  = (__bf16*)alloc((size_t)N1_ * DIN * 2);
  __bf16* X0h  = (__bf16*)alloc((size_t)N1_ * DIN * 2);
  __bf16* X0l  = (__bf16*)alloc((size_t)N1_ * DIN * 2);
  __bf16* Xm0h = (__bf16*)alloc((size_t)P2 * DIN * 2);
  __bf16* Xm0l = (__bf16*)alloc((size_t)P2 * DIN * 2);
  __bf16* Mb0h = (__bf16*)alloc((size_t)P2 * DIN * 2);
  __bf16* Mb0l = (__bf16*)alloc((size_t)P2 * DIN * 2);
  __bf16* M1h  = (__bf16*)alloc((size_t)P2 * DH * 2);
  __bf16* M1l  = (__bf16*)alloc((size_t)P2 * DH * 2);
  __bf16* Mb1h = (__bf16*)alloc((size_t)P2 * DH * 2);
  __bf16* Mb1l = (__bf16*)alloc((size_t)P2 * DH * 2);
  __bf16* Xm1h = (__bf16*)alloc((size_t)P2 * DH * 2);
  __bf16* Xm1l = (__bf16*)alloc((size_t)P2 * DH * 2);
  __bf16* Wl0h = (__bf16*)alloc((size_t)DIN * DH * 2);
  __bf16* Wl0l = (__bf16*)alloc((size_t)DIN * DH * 2);
  __bf16* Wr0h = (__bf16*)alloc((size_t)DIN * DH * 2);
  __bf16* Wr0l = (__bf16*)alloc((size_t)DIN * DH * 2);
  __bf16* Wl1h = (__bf16*)alloc((size_t)DH * DH * 2);
  __bf16* Wl1l = (__bf16*)alloc((size_t)DH * DH * 2);
  __bf16* Wr1h = (__bf16*)alloc((size_t)DH * DH * 2);
  __bf16* Wr1l = (__bf16*)alloc((size_t)DH * DH * 2);

  dim3 B(256);

  // 1) zero sums/counts
  k_zero<<<dim3(4096), B, 0, stream>>>((float*)d_ws, zeroN);

  // 2) mixed hop-0 roots (only first N2 rows are ever consumed)
  k_make_A<<<dim3(2048), B, 0, stream>>>(x0, nullptr, idx, mr,
                                         N2_, DIN, 2, Xm0h, Xm0l);

  // 3) hop-0 segment sums (b-branch rows >= N2 are dead -> skipped)
  k_edge_agg<<<dim3(E0_ / 8), B, 0, stream>>>(x0, src0, dst0, nullptr,
                                              E0_, DIN, N1_, sum0, cnt0);
  k_edge_agg<<<dim3(E0_ / 8), B, 0, stream>>>(x0, srcb0, dstb0, idx,
                                              E0_, DIN, N2_, sumb0, cntb0);

  // 4) means + roots -> bf16x2 fragments
  k_make_A<<<dim3(4096), B, 0, stream>>>(sum0, cnt0, nullptr, nullptr,
                                         N1_, DIN, 1, M0h, M0l);
  k_make_A<<<dim3(2048), B, 0, stream>>>(sumb0, cntb0, nullptr, nullptr,
                                         N2_, DIN, 1, Mb0h, Mb0l);
  k_make_A<<<dim3(4096), B, 0, stream>>>(x0, nullptr, nullptr, nullptr,
                                         N1_, DIN, 0, X0h, X0l);

  // 5) weights -> bf16x2 fragments
  k_make_B<<<dim3(128), B, 0, stream>>>(Wl0, DIN, Wl0h, Wl0l);
  k_make_B<<<dim3(128), B, 0, stream>>>(Wr0, DIN, Wr0h, Wr0l);
  k_make_B<<<dim3(256), B, 0, stream>>>(Wl1, DH, Wl1h, Wl1l);
  k_make_B<<<dim3(256), B, 0, stream>>>(Wr1, DH, Wr1h, Wr1l);

  // 6) hop-0 GEMMs
  //    x1   = relu(mean0@Wl0 + x0@Wr0 + b0)                (all N1 rows)
  //    xm1  = m*relu(mean0@Wl0 + xmix0@Wr0 + b0)
  //         + (1-m)*relu(meanb0@Wl0 + xmix0@Wr0 + b0)      (N2 rows only)
  k_gemm_dual<<<dim3(N1_ / 32), B, 0, stream>>>(
      M0h, M0l, Wl0h, Wl0l, X0h, X0l, Wr0h, Wr0l, b0, x1, N1_, 4, 0, mr, 0);
  k_gemm_dual<<<dim3((N2_ + 31) / 32), B, 0, stream>>>(
      M0h, M0l, Wl0h, Wl0l, Xm0h, Xm0l, Wr0h, Wr0l, b0, xm1, N2_, 4, 1, mr, 0);
  k_gemm_dual<<<dim3((N2_ + 31) / 32), B, 0, stream>>>(
      Mb0h, Mb0l, Wl0h, Wl0l, Xm0h, Xm0l, Wr0h, Wr0l, b0, xm1, N2_, 4, 2, mr, 1);

  // 7) hop-1 segment sums over x1 (b-branch via idx permutation)
  k_edge_agg<<<dim3(E1_ / 8), B, 0, stream>>>(x1, src1, dst1, nullptr,
                                              E1_, DH, N2_, sum1, cnt1);
  k_edge_agg<<<dim3(E1_ / 8), B, 0, stream>>>(x1, srcb1, dstb1, idx,
                                              E1_, DH, N2_, sumb1, cntb1);

  // 8) hop-1 fragments
  k_make_A<<<dim3(4096), B, 0, stream>>>(sum1, cnt1, nullptr, nullptr,
                                         N2_, DH, 1, M1h, M1l);
  k_make_A<<<dim3(4096), B, 0, stream>>>(sumb1, cntb1, nullptr, nullptr,
                                         N2_, DH, 1, Mb1h, Mb1l);
  k_make_A<<<dim3(4096), B, 0, stream>>>(xm1, nullptr, nullptr, nullptr,
                                         N2_, DH, 0, Xm1h, Xm1l);

  // 9) hop-1 GEMMs -> xm2 = m*relu(...) + (1-m)*relu(...)
  k_gemm_dual<<<dim3((N2_ + 31) / 32), B, 0, stream>>>(
      M1h, M1l, Wl1h, Wl1l, Xm1h, Xm1l, Wr1h, Wr1l, b1, xm2, N2_, 8, 1, mr, 0);
  k_gemm_dual<<<dim3((N2_ + 31) / 32), B, 0, stream>>>(
      Mb1h, Mb1l, Wl1h, Wl1l, Xm1h, Xm1l, Wr1h, Wr1l, b1, xm2, N2_, 8, 2, mr, 1);

  // 10) classifier + log_softmax
  k_logits_lsm<<<dim3((N2_ + 7) / 8), B, 0, stream>>>(xm2, Wlin, blin, out,
                                                      N2_);
}
